// SoftmaxNeighborFinder_4458176053364
// MI455X (gfx1250) — matrix operand order repaired and checked by reading, not compiled, not run on hardware
//
#include <hip/hip_runtime.h>

// softmax+top-k is monotonic in the raw dot products -> fused GEMM + running
// top-16 per row. bf16 split (hi+lo, 3 WMMAs per K=32) gives ~f32 accuracy on
// the CDNA5 bf16 matrix pipe. Y (32MB) lives in L2 (192MB) after first pass.

typedef __attribute__((ext_vector_type(16))) __bf16 v16bf;
typedef __attribute__((ext_vector_type(8)))  __bf16 v8bf;
typedef __attribute__((ext_vector_type(8)))  float  v8f;

static constexpr int kNx     = 8192;
static constexpr int kNy     = 16384;
static constexpr int kD      = 512;
static constexpr int kK      = 16;    // neighbors
static constexpr int kMTile  = 64;    // X rows per block
static constexpr int kWaves  = 8;     // 256 threads, wave32
static constexpr int kPitch  = 520;   // bf16 elems per LDS row (512 + 8 pad)
static constexpr int kDPitch = 17;    // f32 dump pitch (16 + 1 pad)

__device__ __forceinline__ void topk_insert(float (&v)[16], int (&ix)[16],
                                            float c, int ci) {
  if (c > v[15]) {                       // usually false -> cheap skip
    v[15] = c; ix[15] = ci;
    #pragma unroll
    for (int p = 15; p > 0; --p) {       // single bubble pass, stays in VGPRs
      if (v[p] > v[p-1]) {
        float tv = v[p]; v[p] = v[p-1]; v[p-1] = tv;
        int   tj = ix[p]; ix[p] = ix[p-1]; ix[p-1] = tj;
      }
    }
  }
}

__global__ __launch_bounds__(256, 1)
void softmax_topk_wmma(const float* __restrict__ X, const float* __restrict__ Y,
                       int* __restrict__ out) {
  extern __shared__ char lds_raw[];
  __bf16* xhi  = (__bf16*)lds_raw;                       // [kMTile][kPitch]
  __bf16* xlo  = xhi + kMTile * kPitch;                  // [kMTile][kPitch]
  float*  dump = (float*)(xlo + kMTile * kPitch);        // [kWaves][kMTile][kDPitch]
  float*  pval = dump + kWaves * kMTile * kDPitch;       // [kWaves][kMTile][kK]
  int*    pidx = (int*)(pval + kWaves * kMTile * kK);    // [kWaves][kMTile][kK]

  const int  tid    = threadIdx.x;
  const int  wave   = tid >> 5;
  const int  lane   = tid & 31;
  const int  row0   = blockIdx.x * kMTile;
  const bool hiHalf = (lane >= 16);
  const int  ncol   = lane & 15;

  // ---- Stage X tile into LDS as split-bf16 (hi + residual lo), once ----
  for (int i = tid; i < kMTile * (kD / 4); i += 256) {
    const int r = i / (kD / 4);
    const int c = (i - r * (kD / 4)) * 4;
    const float4 v4 = *(const float4*)(X + (size_t)(row0 + r) * kD + c);
    const float  f[4] = {v4.x, v4.y, v4.z, v4.w};
    #pragma unroll
    for (int j = 0; j < 4; ++j) {
      const __bf16 h = (__bf16)f[j];
      xhi[r * kPitch + c + j] = h;
      xlo[r * kPitch + c + j] = (__bf16)(f[j] - (float)h);
    }
  }
  __syncthreads();

  // ---- Each lane owns 2 of the block's 64 rows: register-resident top-16 ----
  float tv0[16], tv1[16];
  int   ti0[16], ti1[16];
  #pragma unroll
  for (int j = 0; j < 16; ++j) {
    tv0[j] = tv1[j] = -__builtin_inff();
    ti0[j] = ti1[j] = 0;
  }

  const int aoff = hiHalf ? 8 : 0;    // A: 16-bit layout, K {0..7,16..23} / {8..15,24..31}
  const int boff = hiHalf ? 8 : 0;    // B: 16-bit layout, K {0..7,16..23} / {8..15,24..31}
  float* mydump = dump + wave * kMTile * kDPitch;
  const int r0 = 2 * lane, r1 = 2 * lane + 1;

  for (int nb = wave; nb < kNy / 16; nb += kWaves) {      // waves split the N loop
    const int nbase = nb * 16;
    v8f C[4] = {};
    const float* yrow = Y + (size_t)(nbase + ncol) * kD;  // hot in L2 after pass 1

    for (int kc = 0; kc < kD; kc += 32) {
      // B fragment: two runs of 8 f32 per lane (K boff..+7, boff+16..+23), split into hi/lo bf16 on the fly
      const float4 b0 = *(const float4*)(yrow + kc + boff + 0);
      const float4 b1 = *(const float4*)(yrow + kc + boff + 4);
      const float4 b2 = *(const float4*)(yrow + kc + boff + 16);
      const float4 b3 = *(const float4*)(yrow + kc + boff + 20);
      const float bfv[16] = {b0.x, b0.y, b0.z, b0.w, b1.x, b1.y, b1.z, b1.w,
                             b2.x, b2.y, b2.z, b2.w, b3.x, b3.y, b3.z, b3.w};
      v16bf Bh, Bl;
      #pragma unroll
      for (int j = 0; j < 16; ++j) {
        const __bf16 h = (__bf16)bfv[j];
        Bh[j] = h;
        Bl[j] = (__bf16)(bfv[j] - (float)h);
      }

      #pragma unroll
      for (int s = 0; s < 4; ++s) {   // 4 M-subtiles reuse the B fragments
        const __bf16* ah = xhi + (s * 16 + ncol) * kPitch + kc + aoff;
        const __bf16* al = xlo + (s * 16 + ncol) * kPitch + kc + aoff;
        const v16bf Ah = __builtin_shufflevector(
            *(const v8bf*)ah, *(const v8bf*)(ah + 16),
            0, 1, 2, 3, 4, 5, 6, 7, 8, 9, 10, 11, 12, 13, 14, 15);
        const v16bf Al = __builtin_shufflevector(
            *(const v8bf*)al, *(const v8bf*)(al + 16),
            0, 1, 2, 3, 4, 5, 6, 7, 8, 9, 10, 11, 12, 13, 14, 15);
        // hi*hi + hi*lo + lo*hi  (lo*lo term is below the noise floor)
        C[s] = __builtin_amdgcn_wmma_f32_16x16x32_bf16(false, Ah, false, Bh,
                                                       (short)0, C[s], false, false);
        C[s] = __builtin_amdgcn_wmma_f32_16x16x32_bf16(false, Ah, false, Bl,
                                                       (short)0, C[s], false, false);
        C[s] = __builtin_amdgcn_wmma_f32_16x16x32_bf16(false, Al, false, Bh,
                                                       (short)0, C[s], false, false);
      }
    }

    // ---- Dump C tiles (lane = N, VGPR = M-half) to LDS, then merge per row ----
    #pragma unroll
    for (int s = 0; s < 4; ++s) {
      #pragma unroll
      for (int v = 0; v < 8; ++v) {
        mydump[(s * 16 + (hiHalf ? 8 : 0) + v) * kDPitch + ncol] = C[s][v];
      }
    }
    asm volatile("s_wait_dscnt 0" ::: "memory");  // same-wave LDS RAW fence

    #pragma unroll
    for (int j = 0; j < 16; ++j) {
      topk_insert(tv0, ti0, mydump[r0 * kDPitch + j], nbase + j);
      topk_insert(tv1, ti1, mydump[r1 * kDPitch + j], nbase + j);
    }
  }

  // ---- Publish per-wave sorted partial lists ----
  #pragma unroll
  for (int j = 0; j < 16; ++j) {
    pval[(wave * kMTile + r0) * kK + j] = tv0[j];
    pidx[(wave * kMTile + r0) * kK + j] = ti0[j];
    pval[(wave * kMTile + r1) * kK + j] = tv1[j];
    pidx[(wave * kMTile + r1) * kK + j] = ti1[j];
  }
  __syncthreads();

  // ---- Final cross-wave merge: one thread per row (sorted lists, early break) ----
  if (tid < kMTile) {
    float fv[16];
    int   fi[16];
    #pragma unroll
    for (int j = 0; j < 16; ++j) {
      fv[j] = pval[tid * kK + j];
      fi[j] = pidx[tid * kK + j];
    }
    for (int w = 1; w < kWaves; ++w) {
      for (int j = 0; j < 16; ++j) {
        const float c = pval[(w * kMTile + tid) * kK + j];
        if (c <= fv[15]) break;  // list is descending; rest cannot enter
        topk_insert(fv, fi, c, pidx[(w * kMTile + tid) * kK + j]);
      }
    }
    #pragma unroll
    for (int j = 0; j < 16; ++j) out[(row0 + tid) * kK + j] = fi[j];
  }
}

extern "C" void kernel_launch(void* const* d_in, const int* in_sizes, int n_in,
                              void* d_out, int out_size, void* d_ws, size_t ws_size,
                              hipStream_t stream) {
  (void)in_sizes; (void)n_in; (void)out_size; (void)d_ws; (void)ws_size;
  const float* X = (const float*)d_in[0];   // [8192, 512] f32
  const float* Y = (const float*)d_in[1];   // [16384, 512] f32
  int* out = (int*)d_out;                   // [8192, 16] int32 indices

  const size_t lds_bytes =
      (size_t)2 * kMTile * kPitch * sizeof(__bf16)      // xhi + xlo   (133,120)
    + (size_t)kWaves * kMTile * kDPitch * sizeof(float) // dump        ( 34,816)
    + (size_t)kWaves * kMTile * kK * 8;                 // pval + pidx ( 65,536)
                                                        // total 233,472 B <= 320 KB

  softmax_topk_wmma<<<dim3(kNx / kMTile), dim3(256), lds_bytes, stream>>>(X, Y, out);
}
